// LSTM_18605798326894
// MI455X (gfx1250) — compile-verified, matrix operand order checked by reading, not run on hardware
//
#include <hip/hip_runtime.h>
#include <hip/hip_bf16.h>

typedef __bf16 bf16_t;
typedef __attribute__((ext_vector_type(16))) __bf16 v16bf;
typedef __attribute__((ext_vector_type(8)))  __bf16 v8bf;
typedef __attribute__((ext_vector_type(8)))  float  v8f;

#define B_    16384
#define T_    101
#define I_    4
#define H_    512
#define G4H   2048
#define M_    512
#define ROWS  32      // batch rows per block (2 WMMA M-tiles)
#define NTHR  512     // 16 waves (wave32)

// ---- prep: W_hh f32 -> bf16 (row-major 2048 x 512), stays L2-resident ----
__global__ void whh_to_bf16(const float* __restrict__ w, bf16_t* __restrict__ o, int n) {
  int i = blockIdx.x * blockDim.x + threadIdx.x;
  if (i < n) o[i] = (bf16_t)w[i];
}

// Branch-free activations on the hardware trans pipe (v_exp_f32 / v_rcp_f32).
// sigmoid(x) = 1 / (1 + 2^(-x*log2e));  tanh(x) = 2*sigmoid(2x) - 1.
#define LOG2E 1.4426950408889634f
__device__ __forceinline__ float fast_sigm(float x) {
  float e = __builtin_amdgcn_exp2f(-LOG2E * x);
  return __builtin_amdgcn_rcpf(1.0f + e);
}
__device__ __forceinline__ float fast_tanh(float x) {
  float e = __builtin_amdgcn_exp2f(-2.0f * LOG2E * x);   // e^{-2x}
  return 2.0f * __builtin_amdgcn_rcpf(1.0f + e) - 1.0f;
}

// ---- persistent LSTM: each block owns 32 batch rows for all 101 steps ----
__global__ __launch_bounds__(NTHR, 1)
void lstm_persistent(const float* __restrict__ x_in, const float* __restrict__ a_in,
                     const float* __restrict__ W_ih, const bf16_t* __restrict__ Whh_bf,
                     const float* __restrict__ b_ih, const float* __restrict__ b_hh,
                     const float* __restrict__ W1,   const float* __restrict__ b1,
                     const float* __restrict__ W2,   const float* __restrict__ b2,
                     float* __restrict__ out)
{
  __shared__ float  lds_wih[G4H][4];     // 32 KB
  __shared__ float  lds_bias[G4H];       //  8 KB (b_ih + b_hh)
  __shared__ bf16_t lds_h[ROWS][H_];     // 32 KB  (h state, bf16 for WMMA A)
  __shared__ float  lds_x[ROWS][I_];     // 512 B  (x_t slice)
  __shared__ float  lds_red[ROWS][16];   //  2 KB  (MLP reduction)

  const int tid   = threadIdx.x;
  const int lane  = tid & 31;
  const int wv    = tid >> 5;            // wave 0..15 -> owns hidden cols [32*wv, 32*wv+32)
  const int nlane = lane & 15;           // N column within a 16-tile / M row for A
  const int moff  = (lane < 16) ? 0 : 8; // C/D: lanes 16-31 hold M+8
  const int kbase = (lane < 16) ? 0 : 8; // A/B 16-bit K layout base
  const int rowbase = blockIdx.x * ROWS;

  // stage W_ih + bias sums, zero h
  for (int g = tid; g < G4H; g += NTHR) {
    lds_bias[g]   = b_ih[g] + b_hh[g];
    lds_wih[g][0] = W_ih[g*4 + 0];
    lds_wih[g][1] = W_ih[g*4 + 1];
    lds_wih[g][2] = W_ih[g*4 + 2];
    lds_wih[g][3] = W_ih[g*4 + 3];
  }
  for (int i = tid; i < ROWS * H_; i += NTHR) ((bf16_t*)lds_h)[i] = (bf16_t)0.0f;
  __syncthreads();

  // acc[gate][Mt][u] : 16 fragments of 16x16 f32 ; cfr: cell state (f32, registers)
  v8f acc[4][2][2];
  v8f cfr[2][2];
  #pragma unroll
  for (int Mt = 0; Mt < 2; ++Mt)
    #pragma unroll
    for (int u = 0; u < 2; ++u)
      #pragma unroll
      for (int r = 0; r < 8; ++r) cfr[Mt][u][r] = 0.0f;

  for (int t = 0; t < T_; ++t) {
    // fetch x[:, t, :] slice for our 32 rows
    if (tid < ROWS * I_) {
      int r = tid >> 2, i = tid & 3;
      lds_x[r][i] = x_in[(size_t)(rowbase + r) * (T_ * I_) + t * I_ + i];
    }
    __syncthreads();  // barrier A: x_t ready; prev-step h writes visible

    // accumulator init = x_proj (I=4 folded in) + biases
    #pragma unroll
    for (int g = 0; g < 4; ++g)
      #pragma unroll
      for (int u = 0; u < 2; ++u) {
        const int gc = g * H_ + wv * 32 + u * 16 + nlane;
        const float bs = lds_bias[gc];
        const float w0 = lds_wih[gc][0], w1 = lds_wih[gc][1];
        const float w2 = lds_wih[gc][2], w3 = lds_wih[gc][3];
        #pragma unroll
        for (int Mt = 0; Mt < 2; ++Mt)
          #pragma unroll
          for (int r = 0; r < 8; ++r) {
            const int row = Mt * 16 + r + moff;
            acc[g][Mt][u][r] = bs + lds_x[row][0] * w0 + lds_x[row][1] * w1
                                  + lds_x[row][2] * w2 + lds_x[row][3] * w3;
          }
      }

    // K loop: gates += h @ W_hh^T   (bf16 WMMA, f32 accumulate)
    for (int kk = 0; kk < 16; ++kk) {
      const int k0 = kk * 32;

      // A fragments (h rows) from LDS, ISA 16-bit A layout
      v16bf afr[2];
      #pragma unroll
      for (int Mt = 0; Mt < 2; ++Mt) {
        const bf16_t* hp = &lds_h[Mt * 16 + nlane][k0 + kbase];
        v8bf lo = *(const v8bf*)hp;
        v8bf hi = *(const v8bf*)(hp + 16);
        #pragma unroll
        for (int e = 0; e < 8; ++e) { afr[Mt][e] = lo[e]; afr[Mt][e + 8] = hi[e]; }
      }

      #pragma unroll
      for (int g = 0; g < 4; ++g)
        #pragma unroll
        for (int u = 0; u < 2; ++u) {
          // B fragment: column = W_hh row (L2-resident bf16), same K layout
          const int bc = g * H_ + wv * 32 + u * 16 + nlane;
          const bf16_t* wp = Whh_bf + (size_t)bc * H_ + k0 + kbase;
          v8bf lo = *(const v8bf*)wp;
          v8bf hi = *(const v8bf*)(wp + 16);
          v16bf bfr;
          #pragma unroll
          for (int e = 0; e < 8; ++e) { bfr[e] = lo[e]; bfr[e + 8] = hi[e]; }

          #pragma unroll
          for (int Mt = 0; Mt < 2; ++Mt)
            acc[g][Mt][u] = __builtin_amdgcn_wmma_f32_16x16x32_bf16(
                false, afr[Mt], false, bfr, (short)0, acc[g][Mt][u], false, false);
        }
    }

    // gate nonlinearity + state update (wave-local, branch-free trans-pipe math)
    #pragma unroll
    for (int Mt = 0; Mt < 2; ++Mt)
      #pragma unroll
      for (int u = 0; u < 2; ++u)
        #pragma unroll
        for (int r = 0; r < 8; ++r) {
          const float iv = fast_sigm(acc[0][Mt][u][r]);
          const float fv = fast_sigm(acc[1][Mt][u][r]);
          const float gv = fast_tanh(acc[2][Mt][u][r]);
          const float ov = fast_sigm(acc[3][Mt][u][r]);
          const float cn = fv * cfr[Mt][u][r] + iv * gv;
          cfr[Mt][u][r]  = cn;
          acc[3][Mt][u][r] = ov * fast_tanh(cn);   // reuse acc[3] as h_new
        }

    __syncthreads();  // barrier B: all h reads of this step done

    // publish h_new as bf16 (fragment -> LDS per C/D layout)
    #pragma unroll
    for (int Mt = 0; Mt < 2; ++Mt)
      #pragma unroll
      for (int u = 0; u < 2; ++u) {
        const int col = wv * 32 + u * 16 + nlane;
        #pragma unroll
        for (int r = 0; r < 8; ++r)
          lds_h[Mt * 16 + r + moff][col] = (bf16_t)acc[3][Mt][u][r];
      }
  }
  __syncthreads();

  // ---- MLP head: out = relu(h @ W1^T + b1) . W2[:M] + a*W2[M] + b2 ----
  {
    const int row = tid >> 4;   // batch row 0..31
    const int seg = tid & 15;   // 16 threads per row, 32 m-cols each
    float partial = 0.0f;
    for (int m = seg * 32; m < seg * 32 + 32; ++m) {
      const float* w1r = W1 + (size_t)m * H_;
      float s = b1[m];
      for (int k = 0; k < H_; k += 4) {
        s += (float)lds_h[row][k + 0] * w1r[k + 0]
           + (float)lds_h[row][k + 1] * w1r[k + 1]
           + (float)lds_h[row][k + 2] * w1r[k + 2]
           + (float)lds_h[row][k + 3] * w1r[k + 3];
      }
      s = fmaxf(s, 0.0f);
      partial += s * W2[m];
    }
    lds_red[row][seg] = partial;
  }
  __syncthreads();
  if (tid < ROWS) {  // deterministic fixed-order reduction
    float s = 0.0f;
    #pragma unroll
    for (int j = 0; j < 16; ++j) s += lds_red[tid][j];
    const int b = rowbase + tid;
    out[b] = s + a_in[b] * W2[M_] + b2[0];
  }
}

extern "C" void kernel_launch(void* const* d_in, const int* in_sizes, int n_in,
                              void* d_out, int out_size, void* d_ws, size_t ws_size,
                              hipStream_t stream) {
  (void)in_sizes; (void)n_in; (void)out_size; (void)ws_size;
  const float* batch_x = (const float*)d_in[0];
  const float* batch_a = (const float*)d_in[1];
  const float* W_ih    = (const float*)d_in[2];
  const float* W_hh    = (const float*)d_in[3];
  const float* b_ih    = (const float*)d_in[4];
  const float* b_hh    = (const float*)d_in[5];
  const float* W1      = (const float*)d_in[6];
  const float* b1      = (const float*)d_in[7];
  const float* W2      = (const float*)d_in[8];
  const float* b2      = (const float*)d_in[9];
  float*  out    = (float*)d_out;
  bf16_t* whh_bf = (bf16_t*)d_ws;   // 2 MB

  const int n = G4H * H_;
  whh_to_bf16<<<(n + 255) / 256, 256, 0, stream>>>(W_hh, whh_bf, n);
  lstm_persistent<<<B_ / ROWS, NTHR, 0, stream>>>(batch_x, batch_a, W_ih, whh_bf,
                                                  b_ih, b_hh, W1, b1, W2, b2, out);
}
